// CrossScaleGNN_89300960018889
// MI455X (gfx1250) — matrix operand order repaired and checked by reading, not compiled, not run on hardware
//
#include <hip/hip_runtime.h>
#include <hip/hip_bf16.h>

typedef __attribute__((ext_vector_type(16))) __bf16 v16bf;
typedef __attribute__((ext_vector_type(4)))  __bf16 v4bf;
typedef __attribute__((ext_vector_type(8)))  float  v8f;

#define MT 4   // row-tiles (16 rows each) per wave: 64x16 output per wave

// ---------------------------------------------------------------------------
// Degree / normalization kernels
// ---------------------------------------------------------------------------
__global__ void set_ones_kernel(float* __restrict__ p, int n) {
    int i = blockIdx.x * blockDim.x + threadIdx.x;
    if (i < n) p[i] = 1.0f;            // self-loop contributes 1 to every degree
}

__global__ void deg_accum_kernel(const long long* __restrict__ dst,
                                 float* __restrict__ deg, int e) {
    int i = blockIdx.x * blockDim.x + threadIdx.x;
    if (i < e) atomicAdd(&deg[(int)dst[i]], 1.0f);
}

__global__ void rsqrt_kernel(float* __restrict__ p, int n) {
    int i = blockIdx.x * blockDim.x + threadIdx.x;
    if (i < n) p[i] = rsqrtf(p[i]);    // deg >= 1 always (self loops)
}

// ---------------------------------------------------------------------------
// Streaming fp32 -> bf16 conversion (optionally fused ReLU). n % 4 == 0.
// ---------------------------------------------------------------------------
template<int RELU>
__global__ void cvt_bf16_kernel(const float* __restrict__ in,
                                __bf16* __restrict__ out, size_t n)
{
    size_t i = ((size_t)blockIdx.x * blockDim.x + threadIdx.x) * 4;
    if (i >= n) return;
    float4 x = *(const float4*)(in + i);
    if (RELU) {
        x.x = fmaxf(x.x, 0.f); x.y = fmaxf(x.y, 0.f);
        x.z = fmaxf(x.z, 0.f); x.w = fmaxf(x.w, 0.f);
    }
    v4bf o;
    o[0] = (__bf16)x.x; o[1] = (__bf16)x.y;
    o[2] = (__bf16)x.z; o[3] = (__bf16)x.w;
    *(v4bf*)(out + i) = o;
}

// ---------------------------------------------------------------------------
// Pre-pack W[K,Nc] (fp32 row-major) into WMMA B fragments:
//   Bp[ct][ks][lane][e] = bf16( W[ks*32 + hi*16 + e, ct*16 + lr] )
// One thread per (ct, ks, lane); 16 elements each.
// ---------------------------------------------------------------------------
__global__ void pack_w_kernel(const float* __restrict__ W,
                              __bf16* __restrict__ Bp, int K, int Nc)
{
    int tid = blockIdx.x * blockDim.x + threadIdx.x;
    int total = (Nc / 16) * (K / 32) * 32;
    if (tid >= total) return;
    int lane = tid & 31;
    int rest = tid >> 5;
    int ks = rest % (K / 32);
    int ct = rest / (K / 32);
    int hi = lane >> 4, lr = lane & 15;
    __bf16* out = Bp + (size_t)tid * 16;
    const float* src = W + (size_t)(ks * 32 + hi * 16) * Nc + ct * 16 + lr;
#pragma unroll 4
    for (int e = 0; e < 16; ++e)
        out[e] = (__bf16)src[(size_t)e * Nc];
}

// ---------------------------------------------------------------------------
// WMMA GEMM on bf16 A (row-major) x pre-packed bf16 B (+bias).
// One wave32 per 64x16 output block; 1 B fragment feeds 4 WMMAs per k-step.
// A fragment per lane: 16 contiguous bf16 at k0+hi*16 (k = 16*hi + e)
//   -> two global_load_b128. B fragment: one 32B contiguous v16bf load.
// C/D f32: row = hi*8+v, col = lane&15.
// ---------------------------------------------------------------------------
template<int K, int BIAS>
__global__ void __launch_bounds__(32)
wmma_gemm_pk(const __bf16* __restrict__ A,
             const __bf16* __restrict__ Bp,
             const float* __restrict__ bias, float* __restrict__ Out,
             int M, int Nc)
{
    const int lane = threadIdx.x & 31;
    const int hi   = lane >> 4;
    const int lr   = lane & 15;
    const int row0 = blockIdx.x * (16 * MT);
    const int col  = (blockIdx.y << 4) + lr;

    // Clamped row base pointers: loads always in-bounds (branchless).
    const __bf16* aRow[MT];
#pragma unroll
    for (int t = 0; t < MT; ++t) {
        int r = row0 + t * 16 + lr;
        if (r > M - 1) r = M - 1;
        aRow[t] = A + (size_t)r * K;
    }
    const __bf16* bPtr = Bp + (((size_t)blockIdx.y * (K / 32)) * 32 + lane) * 16;

    v8f acc[MT] = {};

#pragma unroll
    for (int ks = 0; ks < K / 32; ++ks) {
        const int k0 = ks * 32;
        v16bf b = *(const v16bf*)(bPtr + (size_t)ks * 32 * 16);
#pragma unroll
        for (int t = 0; t < MT; ++t) {
            union { v16bf v; uint4 u[2]; } a;
            a.u[0] = *(const uint4*)(aRow[t] + k0 + hi * 16);
            a.u[1] = *(const uint4*)(aRow[t] + k0 + hi * 16 + 8);
            acc[t] = __builtin_amdgcn_wmma_f32_16x16x32_bf16(
                         false, a.v, false, b, (short)0, acc[t], false, false);
        }
    }

    const float bv = BIAS ? bias[col] : 0.0f;
#pragma unroll
    for (int t = 0; t < MT; ++t) {
#pragma unroll
        for (int v = 0; v < 8; ++v) {
            int r = row0 + t * 16 + hi * 8 + v;
            if (r < M) Out[(size_t)r * Nc + col] = acc[t][v] + bv;
        }
    }
}

// ---------------------------------------------------------------------------
// agg[i,f] = bias[f] + xw[i,f] * dinv[i]^2   (self-loop term + bias)
// ---------------------------------------------------------------------------
__global__ void init_self_kernel(const float* __restrict__ xw,
                                 const float* __restrict__ dinv,
                                 const float* __restrict__ bias,
                                 float* __restrict__ agg,
                                 int n, int fshift)
{
    size_t idx = (size_t)blockIdx.x * blockDim.x + threadIdx.x;
    if (idx >= ((size_t)n << fshift)) return;
    int i = (int)(idx >> fshift);
    int f = (int)(idx & ((1u << fshift) - 1));
    float di = dinv[i];
    agg[idx] = bias[f] + xw[idx] * di * di;
}

// ---------------------------------------------------------------------------
// Edge scatter: agg[d] += xw[s]*dinv[s]*dinv[d].
// 8 edges per 256-thread block; 32 lanes/edge; float4 per lane (F=128).
// ---------------------------------------------------------------------------
__global__ void __launch_bounds__(256)
scatter_kernel(const long long* __restrict__ src,
               const long long* __restrict__ dst,
               const float* __restrict__ xw,
               const float* __restrict__ dinv,
               float* __restrict__ agg,
               int e, int F)
{
    int sub  = threadIdx.x >> 5;
    int lane = threadIdx.x & 31;
    int ei   = blockIdx.x * 8 + sub;
    if (ei >= e) return;
    int s = (int)src[ei];
    int d = (int)dst[ei];
    float norm = dinv[s] * dinv[d];
    const float4 x = *(const float4*)(xw + (size_t)s * F + lane * 4);
    float* ap = agg + (size_t)d * F + lane * 4;
    atomicAdd(ap + 0, x.x * norm);
    atomicAdd(ap + 1, x.y * norm);
    atomicAdd(ap + 2, x.z * norm);
    atomicAdd(ap + 3, x.w * norm);
}

// ---------------------------------------------------------------------------
// Gate + blend: one wave32 per node (F=128 -> 4 feats/lane)
// h = relu(agg2); hc = comm[map[i]]; w = sigmoid(h.gW0 + hc.gW1 + gb)
// hf (bf16) = w*h + (1-w)*hc   -- written bf16, feeds classifier GEMM
// ---------------------------------------------------------------------------
__global__ void __launch_bounds__(256)
gate_combine_kernel(const float* __restrict__ agg2,
                    const float* __restrict__ comm,
                    const long long* __restrict__ cmap,
                    const float* __restrict__ gW,
                    const float* __restrict__ gb,
                    __bf16* __restrict__ hf,
                    int n, int F)
{
    int wave = threadIdx.x >> 5;
    int lane = threadIdx.x & 31;
    int node = blockIdx.x * 8 + wave;
    if (node >= n) return;
    int cm = (int)cmap[node];

    float hv[4], hcv[4];
    float partial = 0.0f;
#pragma unroll
    for (int j = 0; j < 4; ++j) {
        int f = lane + 32 * j;
        float h  = fmaxf(agg2[(size_t)node * F + f], 0.0f);
        float hc = comm[(size_t)cm * F + f];
        hv[j] = h; hcv[j] = hc;
        partial += h * gW[f] + hc * gW[F + f];
    }
#pragma unroll
    for (int off = 16; off >= 1; off >>= 1)
        partial += __shfl_xor(partial, off, 32);

    float w = 1.0f / (1.0f + expf(-(partial + gb[0])));
#pragma unroll
    for (int j = 0; j < 4; ++j) {
        int f = lane + 32 * j;
        hf[(size_t)node * F + f] = (__bf16)(w * hv[j] + (1.0f - w) * hcv[j]);
    }
}

// ---------------------------------------------------------------------------
// In-place log_softmax over C<=64 classes: one wave32 per node, 2 elems/lane
// ---------------------------------------------------------------------------
__global__ void __launch_bounds__(256)
log_softmax_kernel(float* __restrict__ out, int n, int C)
{
    int wave = threadIdx.x >> 5;
    int lane = threadIdx.x & 31;
    int node = blockIdx.x * 8 + wave;
    if (node >= n) return;
    float* row = out + (size_t)node * C;

    float a = (lane      < C) ? row[lane]      : -3.0e38f;
    float b = (lane + 32 < C) ? row[lane + 32] : -3.0e38f;

    float m = fmaxf(a, b);
#pragma unroll
    for (int off = 16; off >= 1; off >>= 1)
        m = fmaxf(m, __shfl_xor(m, off, 32));

    float s = 0.0f;
    if (lane      < C) s += expf(a - m);
    if (lane + 32 < C) s += expf(b - m);
#pragma unroll
    for (int off = 16; off >= 1; off >>= 1)
        s += __shfl_xor(s, off, 32);

    float lse = m + logf(s);
    if (lane      < C) row[lane]      = a - lse;
    if (lane + 32 < C) row[lane + 32] = b - lse;
}

// ---------------------------------------------------------------------------
// Launch
// ---------------------------------------------------------------------------
extern "C" void kernel_launch(void* const* d_in, const int* in_sizes, int n_in,
                              void* d_out, int out_size, void* d_ws, size_t ws_size,
                              hipStream_t stream)
{
    const float*     X     = (const float*)d_in[0];      // [N, NFEAT]
    const long long* adj   = (const long long*)d_in[1];  // [2, E] int64
    const float*     commF = (const float*)d_in[2];      // [NCOMM, F]
    const long long* cmap  = (const long long*)d_in[4];  // [N] int64
    const float*     W0    = (const float*)d_in[5];
    const float*     b0    = (const float*)d_in[6];
    const float*     W1    = (const float*)d_in[7];
    const float*     b1    = (const float*)d_in[8];
    const float*     gW    = (const float*)d_in[9];      // [2F, 1]
    const float*     gb    = (const float*)d_in[10];
    const float*     cW    = (const float*)d_in[11];     // [F, C]
    const float*     cb    = (const float*)d_in[12];

    const int F     = in_sizes[6];            // NHID  = 128
    const int NFEAT = in_sizes[5] / F;        // 256
    const int C     = in_sizes[12];           // NCLASS = 64
    const int N     = in_sizes[0] / NFEAT;    // 100000
    const int E     = in_sizes[1] / 2;        // 1600000
    const int FSH   = __builtin_ctz(F);       // 7

    const long long* src = adj;
    const long long* dst = adj + E;

    // Workspace (~128.5 MB):
    //  dinv[N] | R1 (N*NFEAT bf16 == N*F fp32, aliased Xbf/agg) |
    //  R2 (N*F fp32, XW) | R3 (N*F bf16, hidden) | packed weights
    float*  dinv = (float*)d_ws;
    size_t  npad = ((size_t)N + 255) & ~(size_t)255;
    float*  R1f  = dinv + npad;                       // agg (fp32)
    __bf16* R1b  = (__bf16*)R1f;                      // Xbf (bf16), same bytes
    float*  R2   = R1f + (size_t)N * F;               // XW (fp32)
    __bf16* R3   = (__bf16*)(R2 + (size_t)N * F);     // hidden (bf16)
    __bf16* P0   = R3 + (size_t)N * F;                // packed W0
    __bf16* P1   = P0 + (size_t)(F / 16) * (NFEAT / 32) * 512;
    __bf16* P2   = P1 + (size_t)(F / 16) * (F / 32) * 512;

    const int TPB = 256;
    int nbN  = (N + TPB - 1) / TPB;
    int nbE  = (E + TPB - 1) / TPB;
    size_t totNF = (size_t)N * F;
    int nbNF   = (int)((totNF + TPB - 1) / TPB);
    int nbEdge = (E + 7) / 8;
    int nbW    = (N + 7) / 8;
    int rowBlks = (N + 16 * MT - 1) / (16 * MT);

    // 0) weight pre-pack + X conversion (independent)
    int p0t = (F / 16) * (NFEAT / 32) * 32;
    int p1t = (F / 16) * (F / 32) * 32;
    int p2t = (C / 16) * (F / 32) * 32;
    pack_w_kernel<<<(p0t + TPB - 1) / TPB, TPB, 0, stream>>>(W0, P0, NFEAT, F);
    pack_w_kernel<<<(p1t + TPB - 1) / TPB, TPB, 0, stream>>>(W1, P1, F, F);
    pack_w_kernel<<<(p2t + TPB - 1) / TPB, TPB, 0, stream>>>(cW, P2, F, C);
    size_t nX = (size_t)N * NFEAT;
    cvt_bf16_kernel<0><<<(int)((nX / 4 + TPB - 1) / TPB), TPB, 0, stream>>>(X, R1b, nX);

    // 1) symmetric normalization  dinv = 1/sqrt(deg+1)
    set_ones_kernel<<<nbN, TPB, 0, stream>>>(dinv, N);
    deg_accum_kernel<<<nbE, TPB, 0, stream>>>(dst, dinv, E);
    rsqrt_kernel<<<nbN, TPB, 0, stream>>>(dinv, N);

    // 2) layer 0:  XW0 = Xbf @ W0  -> R2
    dim3 g0(rowBlks, F / 16);
    wmma_gemm_pk<256, 0><<<g0, 32, 0, stream>>>(R1b, P0, nullptr, R2, N, F);
    // agg1 = b0 + XW0*dinv^2 + scatter  -> R1f (overwrites Xbf, now dead)
    init_self_kernel<<<nbNF, TPB, 0, stream>>>(R2, dinv, b0, R1f, N, FSH);
    scatter_kernel<<<nbEdge, TPB, 0, stream>>>(src, dst, R2, dinv, R1f, E, F);
    // h1 = bf16(relu(agg1)) -> R3
    cvt_bf16_kernel<1><<<(int)((totNF / 4 + TPB - 1) / TPB), TPB, 0, stream>>>(R1f, R3, totNF);

    // 3) layer 1:  XW1 = h1 @ W1 -> R2 ; agg2 -> R1f ; scatter
    wmma_gemm_pk<128, 0><<<g0, 32, 0, stream>>>(R3, P1, nullptr, R2, N, F);
    init_self_kernel<<<nbNF, TPB, 0, stream>>>(R2, dinv, b1, R1f, N, FSH);
    scatter_kernel<<<nbEdge, TPB, 0, stream>>>(src, dst, R2, dinv, R1f, E, F);

    // 4) gate + blend -> R3 (bf16 h_final); relu folded in
    gate_combine_kernel<<<nbW, TPB, 0, stream>>>(R1f, commF, cmap, gW, gb, R3, N, F);

    // 5) classifier: logits = h_final @ cls_W + cls_b -> d_out
    dim3 g2(rowBlks, C / 16);
    wmma_gemm_pk<128, 1><<<g2, 32, 0, stream>>>(R3, P2, cb, (float*)d_out, N, C);

    // 6) in-place log_softmax
    log_softmax_kernel<<<nbW, TPB, 0, stream>>>((float*)d_out, N, C);
}
